// LLaMACausalSelfAttention_38147899523090
// MI455X (gfx1250) — compile-verified, matrix operand order checked by reading, not run on hardware
//
#include <hip/hip_runtime.h>
#include <hip/hip_bf16.h>
#include <math.h>

#define T_SEQ 2048
#define C_DIM 2048
#define NHEAD 32
#define KVHEAD 8
#define HDIM 64

typedef __bf16 bf16_t;
typedef __attribute__((ext_vector_type(16))) __bf16 bf16x16;
typedef __attribute__((ext_vector_type(8))) float floatx8;

typedef int v4i __attribute__((vector_size(16)));
typedef __attribute__((address_space(3))) v4i l3_v4i;

// ---------- helpers ----------

__device__ __forceinline__ bf16_t f2bf(float f) {
  union { float fv; unsigned uv; } a; a.fv = f;
  unsigned u = a.uv;
  u += 0x7FFFu + ((u >> 16) & 1u);          // round-to-nearest-even
  union { unsigned short sv; bf16_t bv; } b;
  b.sv = (unsigned short)(u >> 16);
  return b.bv;
}

// Async global->LDS 16-byte copy (ASYNCcnt path); falls back to a
// synchronous VGPR round-trip if the builtin is unavailable.
__device__ __forceinline__ void async_copy16(void* ldst, const void* gsrc) {
#if __has_builtin(__builtin_amdgcn_global_load_async_to_lds_b128)
  __builtin_amdgcn_global_load_async_to_lds_b128(
      (v4i*)gsrc,
      (l3_v4i*)(unsigned int)(unsigned long long)ldst, 0, 0);
#else
  *(uint4*)ldst = *(const uint4*)gsrc;
#endif
}

__device__ __forceinline__ void wait_async0() {
#if __has_builtin(__builtin_amdgcn_s_wait_asynccnt)
  __builtin_amdgcn_s_wait_asynccnt(0);
#elif defined(__AMDGCN__)
  asm volatile("s_wait_asynccnt 0x0" ::: "memory");
#endif
}

// k-index base (even) for VGPR v (0..7) and lane-half, per CDNA5 16-bit
// 16x32 A-matrix layout: lanes0-15: V0..3 -> K=0..7, V4..7 -> K=16..23;
// lanes16-31: +8.
__device__ __forceinline__ int kbase_of(int v, int half) {
  return ((v < 4) ? (2 * v) : (2 * v + 8)) + 8 * half;
}

// Build a 16-element bf16 fragment from a contiguous row (A-style layout).
// Works for both A operands (row = M row) and for B operands when the B
// matrix is stored physically transposed (row = N column, contiguous in K).
__device__ __forceinline__ bf16x16 frag_row(const bf16_t* rowp, int half, int koff) {
  bf16x16 f;
#pragma unroll
  for (int v = 0; v < 8; ++v) {
    int kb = koff + kbase_of(v, half);
    f[2 * v] = rowp[kb];
    f[2 * v + 1] = rowp[kb + 1];
  }
  return f;
}

__device__ __forceinline__ floatx8 wmma_bf16(bf16x16 a, bf16x16 b, floatx8 c) {
  return __builtin_amdgcn_wmma_f32_16x16x32_bf16(false, a, false, b, (short)0, c,
                                                 false, false);
}

// ---------- f32 -> bf16 conversion (plain) ----------

__global__ void __launch_bounds__(256)
cvt_f32_bf16(const float* __restrict__ in, bf16_t* __restrict__ out, int n) {
  int i = blockIdx.x * 256 + threadIdx.x;
  if (i < n) out[i] = f2bf(in[i]);
}

// ---------- f32 [K][N] -> bf16 [N][K] transposed conversion ----------
// 32x32 tile per 256-thread block, LDS-tiled for coalescing both ways.

__global__ void __launch_bounds__(256)
cvt_transpose_bf16(const float* __restrict__ in, bf16_t* __restrict__ out,
                   int K, int N) {
  __shared__ float tile[32][33];
  const int k0 = blockIdx.y * 32, n0 = blockIdx.x * 32;
  const int tr = threadIdx.x >> 5;   // 0..7
  const int tc = threadIdx.x & 31;   // 0..31
#pragma unroll
  for (int i = 0; i < 4; ++i)
    tile[tr + i * 8][tc] = in[(size_t)(k0 + tr + i * 8) * N + n0 + tc];
  __syncthreads();
#pragma unroll
  for (int i = 0; i < 4; ++i)
    out[(size_t)(n0 + tr + i * 8) * K + k0 + tc] = f2bf(tile[tc][tr + i * 8]);
}

// ---------- bf16 WMMA GEMM: C[M,N] = A[M,K] * BT[N,K]^T + bias ----------
// 64x128 tile per 256-thread block (8 wave32 waves; each wave: 4 16x16
// tiles sharing one A fragment). Double-buffered LDS; ALL staging is
// async global->LDS DMA (B is pre-transposed, so rows are contiguous).

__global__ void __launch_bounds__(256)
gemm_bf16(const bf16_t* __restrict__ A, const bf16_t* __restrict__ BT,
          const float* __restrict__ bias, float* __restrict__ C,
          int M, int N, int K) {
  __shared__ __align__(16) bf16_t As[2][64][32];     // [buf][m][k]
  __shared__ __align__(16) bf16_t BsT[2][128][32];   // [buf][n][k]

  const int tid = threadIdx.x;
  const int lane = tid & 31, wave = tid >> 5;
  const int half = lane >> 4, ln = lane & 15;
  const int mi = wave & 3;       // 0..3 : 16-row tile
  const int nj = wave >> 2;      // 0..1 : 64-col half
  const int m0 = blockIdx.y * 64, n0 = blockIdx.x * 128;

  const int ar = tid >> 2, ao = (tid & 3) * 8;    // A: 64 rows x 32, 1x16B
  const int br = tid >> 1, bo = (tid & 1) * 16;   // B: 128 rows x 32, 2x16B

  floatx8 zero = {0.f,0.f,0.f,0.f,0.f,0.f,0.f,0.f};
  floatx8 acc[4] = {zero, zero, zero, zero};

  auto stage = [&](int buf, int k0) {
    async_copy16(&As[buf][ar][ao], A + (size_t)(m0 + ar) * K + k0 + ao);
    const bf16_t* brow = BT + (size_t)(n0 + br) * K + k0 + bo;
    async_copy16(&BsT[buf][br][bo], brow);
    async_copy16(&BsT[buf][br][bo + 8], brow + 8);
  };

  stage(0, 0);
  wait_async0();
  __syncthreads();

  int pb = 0;
  for (int k0 = 0; k0 < K; k0 += 32) {
    const int nb = pb ^ 1;
    const bool has_next = (k0 + 32 < K);
    if (has_next) stage(nb, k0 + 32);   // async DMA overlaps the WMMAs below

    bf16x16 a = frag_row(&As[pb][mi * 16 + ln][0], half, 0);
    bf16x16 bfr[4];
#pragma unroll
    for (int t = 0; t < 4; ++t)
      bfr[t] = frag_row(&BsT[pb][nj * 64 + t * 16 + ln][0], half, 0);
#pragma unroll
    for (int t = 0; t < 4; ++t)
      acc[t] = wmma_bf16(a, bfr[t], acc[t]);

    if (has_next) wait_async0();
    __syncthreads();
    pb = nb;
  }

  const int row0 = m0 + mi * 16 + half * 8;
  const int cbase = n0 + nj * 64 + ln;
#pragma unroll
  for (int t = 0; t < 4; ++t) {
    const int c = cbase + t * 16;
    float bv = bias ? bias[c] : 0.f;
#pragma unroll
    for (int r = 0; r < 8; ++r)
      C[(size_t)(row0 + r) * N + c] = acc[t][r] + bv;
  }
}

// ---------- per-head RMSNorm + RoPE, f32 in -> bf16 head-major out ----------
// in: [T, NH*64] f32   out: [NH, T, 64] bf16

__global__ void __launch_bounds__(64)
norm_rope(const float* __restrict__ in, const float* __restrict__ w,
          const float* __restrict__ cosT, const float* __restrict__ sinT,
          bf16_t* __restrict__ out, int NH) {
  const int t = blockIdx.x, h = blockIdx.y, d = threadIdx.x;
  __shared__ float red[64];
  __shared__ float buf[64];
  float x = in[(size_t)t * (NH * HDIM) + h * HDIM + d];
  red[d] = x * x;
  __syncthreads();
#pragma unroll
  for (int st = 32; st > 0; st >>= 1) {
    if (d < st) red[d] += red[d + st];
    __syncthreads();
  }
  float rms = rsqrtf(red[0] * (1.0f / HDIM) + 1e-5f);
  buf[d] = x * rms * w[d];
  __syncthreads();
  int p = d >> 1;
  float c = cosT[t * (HDIM / 2) + p];
  float s = sinT[t * (HDIM / 2) + p];
  float x0 = buf[2 * p], x1 = buf[2 * p + 1];
  float o = (d & 1) ? (x0 * s + x1 * c) : (x0 * c - x1 * s);
  out[((size_t)h * T_SEQ + t) * HDIM + d] = f2bf(o);
}

// ---------- V relayout: [T, KVH*64] f32 -> [KVH, 64, T] bf16 (d-major) ----

__global__ void __launch_bounds__(256)
v_relayout(const float* __restrict__ in, bf16_t* __restrict__ out) {
  int idx = blockIdx.x * 256 + threadIdx.x;     // 2^20 total
  int t = idx & (T_SEQ - 1);
  int d = (idx >> 11) & 63;
  int kv = idx >> 17;                           // /(64*T_SEQ)
  out[idx] = f2bf(in[(size_t)t * (KVHEAD * HDIM) + kv * HDIM + d]);
}

// ---------- flash attention: per (head, 64-query tile) ----------
// Q: [H, T, 64] bf16   K: [KVH, T, 64] bf16   VT: [KVH, 64, T] bf16
// O: [T, H*64] bf16

__global__ void __launch_bounds__(256)
flash_attn(const bf16_t* __restrict__ Q, const bf16_t* __restrict__ Kh,
           const bf16_t* __restrict__ VhT, bf16_t* __restrict__ O) {
  const int qt = blockIdx.x;        // query tile 0..31
  const int h = blockIdx.y;         // head 0..31
  const int kv = h >> 2;            // GQA: rep = 4
  const int tid = threadIdx.x, lane = tid & 31, wave = tid >> 5;
  const int half = lane >> 4, ln = lane & 15;
  const int si = wave & 3, cj = wave >> 2;

  __shared__ __align__(16) bf16_t Kt[64][64];   // [key][d]
  __shared__ __align__(16) bf16_t Vt[64][64];   // [d][key]
  __shared__ __align__(16) bf16_t Pb[64][64];   // P bf16 [qrow][key]
  __shared__ float Sb[64][65];                  // padded f32 scores
  __shared__ float pmax[4][64], psum[4][64];
  __shared__ float mnewS[64], alphaS[64];
  __shared__ float mstate[64], lstate[64], linvS[64];

  const bf16_t* Qbase  = Q + ((size_t)h * T_SEQ + qt * 64) * HDIM;
  const bf16_t* Kbase  = Kh + (size_t)kv * T_SEQ * HDIM;
  const bf16_t* VbaseT = VhT + (size_t)kv * HDIM * T_SEQ;   // [64][T]

  const int m = si * 16 + ln;                   // q row within tile (per lane)
  bf16x16 qa0 = frag_row(Qbase + (size_t)m * HDIM, half, 0);
  bf16x16 qa1 = frag_row(Qbase + (size_t)m * HDIM, half, 32);

  floatx8 o0 = {0.f,0.f,0.f,0.f,0.f,0.f,0.f,0.f};
  floatx8 o1 = {0.f,0.f,0.f,0.f,0.f,0.f,0.f,0.f};

  if (tid < 64) { mstate[tid] = -1e30f; lstate[tid] = 0.f; }

  const int sr = tid >> 2, so = (tid & 3) * 16; // staging: row, 16-elem chunk
  const int srow = tid & 63, sq = tid >> 6, scol0 = sq * 16;

  for (int kt = 0; kt <= qt; ++kt) {
    // --- stage K tile and transposed-V tile, both via async DMA ---
    {
      const bf16_t* krow = Kbase + (size_t)(kt * 64 + sr) * HDIM + so;
      async_copy16(&Kt[sr][so], krow);
      async_copy16(&Kt[sr][so + 8], krow + 8);
      const bf16_t* vrow = VbaseT + (size_t)sr * T_SEQ + kt * 64 + so;
      async_copy16(&Vt[sr][so], vrow);
      async_copy16(&Vt[sr][so + 8], vrow + 8);
      if (kt < qt) {  // prefetch next K/V tile into cache
        __builtin_prefetch(krow + 64 * HDIM, 0, 1);
        __builtin_prefetch(vrow + 64, 0, 1);
      }
    }
    wait_async0();
    __syncthreads();

    // --- S = Q * K^T  (B operand = K rows: A-style layout) ---
    floatx8 s0 = {0.f,0.f,0.f,0.f,0.f,0.f,0.f,0.f};
    floatx8 s1 = {0.f,0.f,0.f,0.f,0.f,0.f,0.f,0.f};
    {
      const bf16_t* kr0 = &Kt[cj * 32 + ln][0];
      const bf16_t* kr1 = &Kt[cj * 32 + 16 + ln][0];
      bf16x16 kb00 = frag_row(kr0, half, 0);
      bf16x16 kb10 = frag_row(kr1, half, 0);
      bf16x16 kb01 = frag_row(kr0, half, 32);
      bf16x16 kb11 = frag_row(kr1, half, 32);
      s0 = wmma_bf16(qa0, kb00, s0);
      s1 = wmma_bf16(qa0, kb10, s1);
      s0 = wmma_bf16(qa1, kb01, s0);
      s1 = wmma_bf16(qa1, kb11, s1);
    }
    {
      int r0 = si * 16 + half * 8;
      int c0 = cj * 32 + ln;
#pragma unroll
      for (int r = 0; r < 8; ++r) {
        Sb[r0 + r][c0]      = s0[r];
        Sb[r0 + r][c0 + 16] = s1[r];
      }
    }
    __syncthreads();

    // --- online softmax: 4 lanes per query row, 16 cols each ---
    {
      float mt = -1e30f;
#pragma unroll
      for (int j = 0; j < 16; ++j) {
        int col = scol0 + j;
        float sv = Sb[srow][col] * 0.125f;            // 1/sqrt(64)
        if (kt * 64 + col > qt * 64 + srow) sv = -1e30f;  // causal mask
        Sb[srow][col] = sv;
        mt = fmaxf(mt, sv);
      }
      pmax[sq][srow] = mt;
    }
    __syncthreads();
    if (tid < 64) {
      float m4 = fmaxf(fmaxf(pmax[0][tid], pmax[1][tid]),
                       fmaxf(pmax[2][tid], pmax[3][tid]));
      float mold = mstate[tid];
      float mnew = fmaxf(mold, m4);
      mstate[tid] = mnew;
      mnewS[tid] = mnew;
      alphaS[tid] = __expf(mold - mnew);
    }
    __syncthreads();
    {
      float mnew = mnewS[srow];
      float ls = 0.f;
#pragma unroll
      for (int j = 0; j < 16; ++j) {
        int col = scol0 + j;
        float pv = __expf(Sb[srow][col] - mnew);
        ls += pv;
        Pb[srow][col] = f2bf(pv);
      }
      psum[sq][srow] = ls;
    }
    __syncthreads();
    if (tid < 64) {
      float ls = psum[0][tid] + psum[1][tid] + psum[2][tid] + psum[3][tid];
      lstate[tid] = lstate[tid] * alphaS[tid] + ls;
    }

    // --- O = O*alpha + P * V  (B operand = Vt rows: A-style layout) ---
    {
      int r0 = si * 16 + half * 8;
#pragma unroll
      for (int r = 0; r < 8; ++r) {
        float al = alphaS[r0 + r];
        o0[r] *= al;
        o1[r] *= al;
      }
      const bf16_t* vr0 = &Vt[cj * 32 + ln][0];
      const bf16_t* vr1 = &Vt[cj * 32 + 16 + ln][0];
      bf16x16 pa0 = frag_row(&Pb[m][0], half, 0);
      bf16x16 pa1 = frag_row(&Pb[m][0], half, 32);
      bf16x16 vb00 = frag_row(vr0, half, 0);
      bf16x16 vb10 = frag_row(vr1, half, 0);
      bf16x16 vb01 = frag_row(vr0, half, 32);
      bf16x16 vb11 = frag_row(vr1, half, 32);
      o0 = wmma_bf16(pa0, vb00, o0);
      o1 = wmma_bf16(pa0, vb10, o1);
      o0 = wmma_bf16(pa1, vb01, o0);
      o1 = wmma_bf16(pa1, vb11, o1);
    }
    __syncthreads();
  }

  if (tid < 64) linvS[tid] = 1.0f / lstate[tid];
  __syncthreads();
  {
    int r0 = si * 16 + half * 8;
    int c0 = h * HDIM + cj * 32 + ln;
#pragma unroll
    for (int r = 0; r < 8; ++r) {
      float li = linvS[r0 + r];
      size_t grow = (size_t)(qt * 64 + r0 + r);
      O[grow * C_DIM + c0]      = f2bf(o0[r] * li);
      O[grow * C_DIM + c0 + 16] = f2bf(o1[r] * li);
    }
  }
}

// ---------- host ----------

extern "C" void kernel_launch(void* const* d_in, const int* in_sizes, int n_in,
                              void* d_out, int out_size, void* d_ws, size_t ws_size,
                              hipStream_t stream) {
  (void)in_sizes; (void)n_in; (void)out_size; (void)ws_size;

  const float* x    = (const float*)d_in[0];
  const float* fcos = (const float*)d_in[1];
  const float* fsin = (const float*)d_in[2];
  // d_in[3] = additive mask (unused; causal mask applied analytically)
  const float* wq   = (const float*)d_in[4];
  const float* wq_b = (const float*)d_in[5];
  const float* wk   = (const float*)d_in[6];
  const float* wk_b = (const float*)d_in[7];
  const float* wv   = (const float*)d_in[8];
  const float* wv_b = (const float*)d_in[9];
  const float* wo   = (const float*)d_in[10];
  const float* wo_b = (const float*)d_in[11];
  const float* qn_w = (const float*)d_in[12];
  const float* kn_w = (const float*)d_in[13];
  float* out = (float*)d_out;

  const int CC = T_SEQ * C_DIM;          // 4,194,304
  const int CK = T_SEQ * KVHEAD * HDIM;  // 1,048,576
  const int KVN = KVHEAD * HDIM;         // 512

  char* p = (char*)d_ws;
  auto alloc = [&](size_t bytes) {
    char* r = p;
    p += (bytes + 255) & ~(size_t)255;
    return r;
  };
  bf16_t* xbf  = (bf16_t*)alloc((size_t)CC * 2);
  bf16_t* wqT  = (bf16_t*)alloc((size_t)CC * 2);   // [N=2048][K=2048]
  bf16_t* wkT  = (bf16_t*)alloc((size_t)CK * 2);   // [N=512][K=2048]
  bf16_t* wvT  = (bf16_t*)alloc((size_t)CK * 2);
  bf16_t* woT  = (bf16_t*)alloc((size_t)CC * 2);
  float*  qf   = (float*) alloc((size_t)CC * 4);
  float*  kf   = (float*) alloc((size_t)CK * 4);
  float*  vf   = (float*) alloc((size_t)CK * 4);
  bf16_t* Qbf  = (bf16_t*)alloc((size_t)CC * 2);
  bf16_t* Kbf  = (bf16_t*)alloc((size_t)CK * 2);
  bf16_t* VbfT = (bf16_t*)alloc((size_t)CK * 2);
  bf16_t* abf  = (bf16_t*)alloc((size_t)CC * 2);

  cvt_f32_bf16<<<(CC + 255) / 256, 256, 0, stream>>>(x, xbf, CC);
  cvt_transpose_bf16<<<dim3(C_DIM / 32, C_DIM / 32), 256, 0, stream>>>(
      wq, wqT, C_DIM, C_DIM);
  cvt_transpose_bf16<<<dim3(KVN / 32, C_DIM / 32), 256, 0, stream>>>(
      wk, wkT, C_DIM, KVN);
  cvt_transpose_bf16<<<dim3(KVN / 32, C_DIM / 32), 256, 0, stream>>>(
      wv, wvT, C_DIM, KVN);
  cvt_transpose_bf16<<<dim3(C_DIM / 32, C_DIM / 32), 256, 0, stream>>>(
      wo, woT, C_DIM, C_DIM);

  gemm_bf16<<<dim3(C_DIM / 128, T_SEQ / 64), 256, 0, stream>>>(
      xbf, wqT, wq_b, qf, T_SEQ, C_DIM, C_DIM);
  gemm_bf16<<<dim3(KVN / 128, T_SEQ / 64), 256, 0, stream>>>(
      xbf, wkT, wk_b, kf, T_SEQ, KVN, C_DIM);
  gemm_bf16<<<dim3(KVN / 128, T_SEQ / 64), 256, 0, stream>>>(
      xbf, wvT, wv_b, vf, T_SEQ, KVN, C_DIM);

  norm_rope<<<dim3(T_SEQ, NHEAD), 64, 0, stream>>>(qf, qn_w, fcos, fsin, Qbf, NHEAD);
  norm_rope<<<dim3(T_SEQ, KVHEAD), 64, 0, stream>>>(kf, kn_w, fcos, fsin, Kbf, KVHEAD);
  v_relayout<<<CK / 256, 256, 0, stream>>>(vf, VbfT);

  flash_attn<<<dim3(T_SEQ / 64, NHEAD), 256, 0, stream>>>(Qbf, Kbf, VbfT, abf);

  gemm_bf16<<<dim3(C_DIM / 128, T_SEQ / 64), 256, 0, stream>>>(
      abf, woT, wo_b, out, T_SEQ, C_DIM, C_DIM);
}
